// AutoEncoder_BNN_14396730376924
// MI455X (gfx1250) — matrix operand order repaired and checked by reading, not compiled, not run on hardware
//
#include <hip/hip_runtime.h>

typedef __attribute__((ext_vector_type(16))) __bf16 v16bf;
typedef __attribute__((ext_vector_type(8)))  float  v8f;

#define BM 128
#define BN 128
#define BK 32
#define LDSS 40   // LDS row stride in elements (32 + 8 pad)

__device__ __forceinline__ unsigned short f32_to_bf16(float f) {
  unsigned u = __float_as_uint(f);
  u += 0x7fffu + ((u >> 16) & 1u);   // round-to-nearest-even
  return (unsigned short)(u >> 16);
}

// CDNA5 async copy: global -> LDS directly, tracked by ASYNCcnt.
// ldsoff = low 32 bits of flat LDS address (== LDS byte offset per ISA aperture rule)
__device__ __forceinline__ void async_copy_b128(unsigned ldsoff, const void* gaddr) {
  asm volatile("global_load_async_to_lds_b128 %0, %1, off"
               :: "v"(ldsoff), "v"(gaddr) : "memory");
}
__device__ __forceinline__ void wait_asynccnt0() {
  asm volatile("s_wait_asynccnt 0" ::: "memory");
}

// ---------------- f32 -> bf16 conversion (8 elems / thread) ----------------
__global__ void cvt_f32_bf16(const float* __restrict__ in,
                             unsigned short* __restrict__ out, long n) {
  long i = ((long)blockIdx.x * blockDim.x + threadIdx.x) * 8;
  if (i >= n) return;
  float4 a = *(const float4*)(in + i);
  float4 b = *(const float4*)(in + i + 4);
  uint4 o;
  o.x = (unsigned)f32_to_bf16(a.x) | ((unsigned)f32_to_bf16(a.y) << 16);
  o.y = (unsigned)f32_to_bf16(a.z) | ((unsigned)f32_to_bf16(a.w) << 16);
  o.z = (unsigned)f32_to_bf16(b.x) | ((unsigned)f32_to_bf16(b.y) << 16);
  o.w = (unsigned)f32_to_bf16(b.z) | ((unsigned)f32_to_bf16(b.w) << 16);
  *(uint4*)(out + i) = o;
}

// ---------------- tiled bf16 WMMA GEMM:  C = act(A @ B^T + bias) ----------
// A: M x K (row major, bf16), B: N x K (row major, bf16)  -> "NT" gemm
// grid = (N/BN, M/BM), block = 256 threads = 8 waves (wave32)
// Double-buffered LDS filled by GLOBAL_LOAD_ASYNC_TO_LDS_B128.
__global__ __launch_bounds__(256)
void gemm_bf16_wmma(const unsigned short* __restrict__ A, int lda,
                    const unsigned short* __restrict__ B, int ldb,
                    int K,
                    const float* __restrict__ bias, int relu,
                    float* __restrict__ Cf, int ldcf,
                    unsigned short* __restrict__ Cb, int ldcb) {
  __shared__ __align__(16) unsigned short Alds[2][BM * LDSS];
  __shared__ __align__(16) unsigned short Blds[2][BN * LDSS];

  const int tid  = threadIdx.x;
  const int lane = tid & 31;
  const int wv   = tid >> 5;
  const int wm   = wv & 3;    // wave row slice: 4 x 32 rows
  const int wn   = wv >> 2;   // wave col slice: 2 x 64 cols
  const int hi   = lane >> 4; // lane half (ISA WMMA layout)
  const int lo   = lane & 15;

  const long rowBlock = (long)blockIdx.y * BM;
  const long colBlock = (long)blockIdx.x * BN;

  v8f acc[2][4];
#pragma unroll
  for (int i = 0; i < 2; ++i)
#pragma unroll
    for (int j = 0; j < 4; ++j)
#pragma unroll
      for (int e = 0; e < 8; ++e) acc[i][j][e] = 0.0f;

  // 512 16-byte chunks per 128x32 bf16 tile; 256 threads x 2 chunks each
  const int r0 = tid >> 2,         q0 = tid & 3;
  const int r1 = (tid + 256) >> 2, q1 = (tid + 256) & 3;

  // per-thread global source pointers (advance by BK elements per k-tile)
  const unsigned short* gA0 = A + (rowBlock + r0) * lda + q0 * 8;
  const unsigned short* gA1 = A + (rowBlock + r1) * lda + q1 * 8;
  const unsigned short* gB0 = B + (colBlock + r0) * ldb + q0 * 8;
  const unsigned short* gB1 = B + (colBlock + r1) * ldb + q1 * 8;

  // per-thread LDS destination offsets, both buffers (ptr-to-int also makes
  // the shared arrays escape so the async writes are not optimized around)
  unsigned la0[2], la1[2], lb0[2], lb1[2];
#pragma unroll
  for (int b = 0; b < 2; ++b) {
    la0[b] = (unsigned)(size_t)&Alds[b][r0 * LDSS + q0 * 8];
    la1[b] = (unsigned)(size_t)&Alds[b][r1 * LDSS + q1 * 8];
    lb0[b] = (unsigned)(size_t)&Blds[b][r0 * LDSS + q0 * 8];
    lb1[b] = (unsigned)(size_t)&Blds[b][r1 * LDSS + q1 * 8];
  }

  auto issue_tile = [&](int buf, int k0) {
    async_copy_b128(la0[buf], gA0 + k0);
    async_copy_b128(la1[buf], gA1 + k0);
    async_copy_b128(lb0[buf], gB0 + k0);
    async_copy_b128(lb1[buf], gB1 + k0);
  };

  const int NT = K / BK;
  issue_tile(0, 0);
  wait_asynccnt0();
  __syncthreads();

  for (int t = 0; t < NT; ++t) {
    const int cur = t & 1;
    if (t + 1 < NT) issue_tile((t + 1) & 1, (t + 1) * BK);  // prefetch next

    const unsigned short* Ac = Alds[cur];
    const unsigned short* Bc = Blds[cur];

    union Frag { uint4 u[2]; v16bf v; };
    Frag af[2], bf[4];
#pragma unroll
    for (int fm = 0; fm < 2; ++fm) {
      // A 16x32 bf16 frag: lane lo = row M; elems 0-7: K=hi*8.., elems 8-15: K=16+hi*8..
      const unsigned short* p = Ac + (wm * 32 + fm * 16 + lo) * LDSS;
      af[fm].u[0] = *(const uint4*)(p + hi * 8);
      af[fm].u[1] = *(const uint4*)(p + 16 + hi * 8);
    }
#pragma unroll
    for (int fn = 0; fn < 4; ++fn) {
      // B 32x16 bf16 frag: lane lo = col N; elems 0-7: K=hi*8.., elems 8-15: K=16+hi*8..
      const unsigned short* p = Bc + (wn * 64 + fn * 16 + lo) * LDSS;
      bf[fn].u[0] = *(const uint4*)(p + hi * 8);
      bf[fn].u[1] = *(const uint4*)(p + 16 + hi * 8);
    }
#pragma unroll
    for (int fm = 0; fm < 2; ++fm)
#pragma unroll
      for (int fn = 0; fn < 4; ++fn)
        acc[fm][fn] = __builtin_amdgcn_wmma_f32_16x16x32_bf16(
            false, af[fm].v, false, bf[fn].v, (short)0, acc[fm][fn],
            false, false);

    wait_asynccnt0();   // prefetch into other buffer has landed
    __syncthreads();    // everyone done reading cur before it is refilled
  }

  // Epilogue. C frag layout: VGPR e -> M = e + hi*8, N = lo (within 16x16 tile)
#pragma unroll
  for (int fm = 0; fm < 2; ++fm) {
#pragma unroll
    for (int fn = 0; fn < 4; ++fn) {
      const long col = colBlock + wn * 64 + fn * 16 + lo;
      const float bv = bias ? bias[col] : 0.0f;
      const long rowb = rowBlock + wm * 32 + fm * 16 + hi * 8;
#pragma unroll
      for (int e = 0; e < 8; ++e) {
        float v = acc[fm][fn][e] + bv;
        if (relu) v = fmaxf(v, 0.0f);
        const long r = rowb + e;
        if (Cf) Cf[r * (long)ldcf + col] = v;
        if (Cb) Cb[r * (long)ldcb + col] = f32_to_bf16(v);
      }
    }
  }
}

// ---------------- elementwise p/q combine -> bf16 -------------------------
__global__ void pq_kernel(const float* __restrict__ uw,  // batch x 2*nbus (f32)
                          const float* __restrict__ Gu, const float* __restrict__ Gw,
                          const float* __restrict__ Bu, const float* __restrict__ Bw,
                          const float* __restrict__ bias_p,
                          const float* __restrict__ bias_q,
                          unsigned short* __restrict__ pb,
                          unsigned short* __restrict__ qb,
                          long total, int nbus) {
  long idx = (long)blockIdx.x * blockDim.x + threadIdx.x;
  if (idx >= total) return;
  long r = idx / nbus;
  int  c = (int)(idx - r * nbus);
  float u = uw[r * (2L * nbus) + c];
  float w = uw[r * (2L * nbus) + nbus + c];
  float gu = Gu[idx], gw = Gw[idx], bu = Bu[idx], bw = Bw[idx];
  float p = u * gu + w * gw + w * bu - u * bw + bias_p[c];
  float q = w * gu - u * gw - u * bu - w * bw + bias_q[c];
  pb[idx] = f32_to_bf16(p);
  qb[idx] = f32_to_bf16(q);
}

// ---------------------------------------------------------------------------
extern "C" void kernel_launch(void* const* d_in, const int* in_sizes, int n_in,
                              void* d_out, int out_size, void* d_ws, size_t ws_size,
                              hipStream_t stream) {
  const int BATCH = 8192, NX = 4096, NR = 1024, NB = 2048;

  const float* x      = (const float*)d_in[0];
  const float* W_h    = (const float*)d_in[1];
  const float* b_h    = (const float*)d_in[2];
  const float* W_h2   = (const float*)d_in[3];
  const float* b_h2   = (const float*)d_in[4];
  const float* W_y    = (const float*)d_in[5];
  const float* b_y    = (const float*)d_in[6];
  const float* G      = (const float*)d_in[7];
  const float* Bm     = (const float*)d_in[8];
  const float* bias_p = (const float*)d_in[9];
  const float* bias_q = (const float*)d_in[10];
  const float* W_p    = (const float*)d_in[11];
  const float* b_p    = (const float*)d_in[12];
  const float* W_q    = (const float*)d_in[13];
  const float* b_q    = (const float*)d_in[14];

  float* out_uw = (float*)d_out;                              // 8192 x 4096
  float* out_p  = (float*)d_out + (long)BATCH * NX;           // 8192 x 2048
  float* out_q  = out_p + (long)BATCH * NB;                   // 8192 x 2048

  // ---- workspace layout (bytes) ----
  size_t off = 0;
  auto alloc = [&](size_t bytes) {
    size_t o = off;
    off += (bytes + 255) & ~(size_t)255;
    return (char*)d_ws + o;
  };
  unsigned short* xb   = (unsigned short*)alloc((size_t)BATCH * NX * 2);
  unsigned short* Whb  = (unsigned short*)alloc((size_t)NR * NX * 2);
  unsigned short* Wh2b = (unsigned short*)alloc((size_t)NR * NR * 2);
  unsigned short* Wyb  = (unsigned short*)alloc((size_t)2 * NB * NR * 2);
  unsigned short* Gb   = (unsigned short*)alloc((size_t)NB * NB * 2);
  unsigned short* Bmb  = (unsigned short*)alloc((size_t)NB * NB * 2);
  unsigned short* Wpb  = (unsigned short*)alloc((size_t)NB * NB * 2);
  unsigned short* Wqb  = (unsigned short*)alloc((size_t)NB * NB * 2);
  unsigned short* h1b  = (unsigned short*)alloc((size_t)BATCH * NR * 2);
  unsigned short* h2b  = (unsigned short*)alloc((size_t)BATCH * NR * 2);
  unsigned short* uwb  = (unsigned short*)alloc((size_t)BATCH * NX * 2);
  float*          Gu   = (float*)alloc((size_t)BATCH * NB * 4);
  float*          Gw   = (float*)alloc((size_t)BATCH * NB * 4);
  float*          Bu   = (float*)alloc((size_t)BATCH * NB * 4);
  float*          Bw   = (float*)alloc((size_t)BATCH * NB * 4);
  unsigned short* ppb  = (unsigned short*)alloc((size_t)BATCH * NB * 2);
  unsigned short* qqb  = (unsigned short*)alloc((size_t)BATCH * NB * 2);

  // ---- f32 -> bf16 conversions ----
  auto cvt = [&](const float* src, unsigned short* dst, long n) {
    long threads = n / 8;
    cvt_f32_bf16<<<(unsigned)((threads + 255) / 256), 256, 0, stream>>>(src, dst, n);
  };
  cvt(x,    xb,   (long)BATCH * NX);
  cvt(W_h,  Whb,  (long)NR * NX);
  cvt(W_h2, Wh2b, (long)NR * NR);
  cvt(W_y,  Wyb,  (long)2 * NB * NR);
  cvt(G,    Gb,   (long)NB * NB);
  cvt(Bm,   Bmb,  (long)NB * NB);
  cvt(W_p,  Wpb,  (long)NB * NB);
  cvt(W_q,  Wqb,  (long)NB * NB);

  auto gemm = [&](const unsigned short* A, int lda, const unsigned short* B, int ldb,
                  int M, int N, int K, const float* bias, int relu,
                  float* Cf, int ldcf, unsigned short* Cb, int ldcb) {
    dim3 grid(N / BN, M / BM);
    gemm_bf16_wmma<<<grid, 256, 0, stream>>>(A, lda, B, ldb, K, bias, relu,
                                             Cf, ldcf, Cb, ldcb);
  };

  // h1 = relu(x @ W_h^T + b_h)                     (8192 x 1024)
  gemm(xb, NX, Whb, NX, BATCH, NR, NX, b_h, 1, nullptr, 0, h1b, NR);
  // h2 = relu(h1 @ W_h2^T + b_h2)                  (8192 x 1024)
  gemm(h1b, NR, Wh2b, NR, BATCH, NR, NR, b_h2, 1, nullptr, 0, h2b, NR);
  // uw = h2 @ W_y^T + b_y    -> f32 output + bf16 operand   (8192 x 4096)
  gemm(h2b, NR, Wyb, NR, BATCH, 2 * NB, NR, b_y, 0, out_uw, NX, uwb, NX);

  // Gu/Gw/Bu/Bw (8192 x 2048 each); u = uwb[:, :2048], w = uwb[:, 2048:]
  gemm(uwb,       NX, Gb,  NB, BATCH, NB, NB, nullptr, 0, Gu, NB, nullptr, 0);
  gemm(uwb + NB,  NX, Gb,  NB, BATCH, NB, NB, nullptr, 0, Gw, NB, nullptr, 0);
  gemm(uwb,       NX, Bmb, NB, BATCH, NB, NB, nullptr, 0, Bu, NB, nullptr, 0);
  gemm(uwb + NB,  NX, Bmb, NB, BATCH, NB, NB, nullptr, 0, Bw, NB, nullptr, 0);

  // elementwise combine -> p_pre/q_pre (bf16)
  {
    long total = (long)BATCH * NB;
    pq_kernel<<<(unsigned)((total + 255) / 256), 256, 0, stream>>>(
        out_uw, Gu, Gw, Bu, Bw, bias_p, bias_q, ppb, qqb, total, NB);
  }

  // p = p_pre @ W_p^T + b_p ; q = q_pre @ W_q^T + b_q   (8192 x 2048 f32 out)
  gemm(ppb, NB, Wpb, NB, BATCH, NB, NB, b_p, 0, out_p, NB, nullptr, 0);
  gemm(qqb, NB, Wqb, NB, BATCH, NB, NB, b_q, 0, out_q, NB, nullptr, 0);
}
